// CartBasisStressHead_70712341561540
// MI455X (gfx1250) — matrix operand order repaired and checked by reading, not compiled, not run on hardware
//
#include <hip/hip_runtime.h>

typedef __attribute__((ext_vector_type(16))) _Float16 v16h;
typedef __attribute__((ext_vector_type(8)))  _Float16 v8h;
typedef __attribute__((ext_vector_type(4)))  _Float16 v4h;
typedef __attribute__((ext_vector_type(8)))  float    v8f;
typedef __attribute__((ext_vector_type(4)))  float    v4f;

// Fast SiLU: x * rcp(1 + exp(-x)).  v_exp_f32 + v_rcp_f32, no IEEE divide.
static __device__ __forceinline__ float silu_f(float x) {
    return x * __builtin_amdgcn_rcpf(1.0f + __expf(-x));
}

__global__ void zero_ws_kernel(float* ws, int n) {
    int i = blockIdx.x * blockDim.x + threadIdx.x;
    if (i < n) ws[i] = 0.0f;
}

// One block = 256 threads = 8 waves; each wave handles 16 nodes.
// Fused: 2-layer f16-WMMA MLP (+ scalar head folded into layer-2 epilogue)
// + l2 dot + atomic segment sums.
__global__ __launch_bounds__(256)
void mlp_reduce_kernel(const float* __restrict__ ne,     // (N,9,128)
                       const float* __restrict__ W1,     // (128,128) [H,C]
                       const float* __restrict__ b1,     // (128)
                       const float* __restrict__ W2,     // (128,128)
                       const float* __restrict__ b2,     // (128)
                       const float* __restrict__ W3,     // (1,128)
                       const float* __restrict__ b3,     // (1)
                       const float* __restrict__ wl2g,   // (1,128)
                       const int*   __restrict__ batch,  // (N)
                       float* __restrict__ iso,          // (G)
                       float* __restrict__ aniso,        // (G,5)
                       int N)
{
    __shared__ __attribute__((aligned(32))) _Float16 W1h[128 * 128];
    __shared__ __attribute__((aligned(32))) _Float16 W2h[128 * 128];
    __shared__ __attribute__((aligned(32))) _Float16 hscr[8][16 * 128];
    __shared__ float b1s[128], b2s[128], w3s[128];

    const int tid = threadIdx.x;

    // Stage weights into LDS as f16: float4 loads -> packed 64-bit LDS stores.
    for (int i = tid * 4; i < 128 * 128; i += 256 * 4) {
        v4f w1 = *(const v4f*)(W1 + i);
        v4f w2 = *(const v4f*)(W2 + i);
        v4h h1, h2;
        #pragma unroll
        for (int j = 0; j < 4; ++j) {
            h1[j] = (_Float16)w1[j];
            h2[j] = (_Float16)w2[j];
        }
        *(v4h*)(W1h + i) = h1;
        *(v4h*)(W2h + i) = h2;
    }
    if (tid < 128) {
        b1s[tid] = b1[tid];
        b2s[tid] = b2[tid];
        w3s[tid] = W3[tid];
    }
    __syncthreads();

    const int wave = tid >> 5;
    const int lane = tid & 31;
    const int half = lane >> 4;   // which 16-lane half
    const int lr   = lane & 15;
    const int nodeBase = blockIdx.x * 128 + wave * 16;
    _Float16* hs = &hscr[wave][0];
    const float b3v = b3[0];

    // ---------------- Layer 1: A fragments from global x0 ----------------
    // A 16x32 in contiguous k order (this code's choice, matched by B): lanes 0-15 -> M=lr, K = k0..k0+15;
    // lanes 16-31 -> M=lr, K = k0+16..k0+31.
    v16h a1[4];
    {
        const int  row = nodeBase + lr;
        const bool ok  = row < N;
        const float* xr = ne + (size_t)row * (9 * 128); // x0 row (m==0)
        #pragma unroll
        for (int kk = 0; kk < 4; ++kk) {
            v16h a;
            if (ok) {
                const int c0 = kk * 32 + half * 16;
                v4f p0 = *(const v4f*)(xr + c0);
                v4f p1 = *(const v4f*)(xr + c0 + 4);
                v4f p2 = *(const v4f*)(xr + c0 + 8);
                v4f p3 = *(const v4f*)(xr + c0 + 12);
                #pragma unroll
                for (int i = 0; i < 4; ++i) {
                    a[i]      = (_Float16)p0[i];
                    a[4 + i]  = (_Float16)p1[i];
                    a[8 + i]  = (_Float16)p2[i];
                    a[12 + i] = (_Float16)p3[i];
                }
            } else {
                #pragma unroll
                for (int i = 0; i < 16; ++i) a[i] = (_Float16)0.0f;
            }
            a1[kk] = a;
        }
    }

    // h1 = silu(x0 @ W1^T + b1): 8 column tiles x 4 K-steps of WMMA.
    // B 32x16 in the same contiguous k order as A: lane holds column j = ct*16+lr; lanes 0-15 K=k0..k0+15,
    // lanes 16-31 K=k0+16..k0+31 -> contiguous 32B from W1h[j*128 + k0 + half*16].
    #pragma unroll
    for (int ct = 0; ct < 8; ++ct) {
        const float bias = b1s[ct * 16 + lr];
        v8f acc;
        #pragma unroll
        for (int i = 0; i < 8; ++i) acc[i] = bias;
        const _Float16* wb = &W1h[(ct * 16 + lr) * 128 + half * 16];
        #pragma unroll
        for (int kk = 0; kk < 4; ++kk) {
            v16h b = *(const v16h*)(wb + kk * 32);
            acc = __builtin_amdgcn_wmma_f32_16x16x32_f16(
                false, a1[kk], false, b, (short)0, acc, false, false);
        }
        // C/D layout: VGPR r -> row (r + 8*half), lane lr -> col ct*16+lr
        #pragma unroll
        for (int r = 0; r < 8; ++r)
            hs[(r + 8 * half) * 128 + ct * 16 + lr] = (_Float16)silu_f(acc[r]);
    }

    // ---------------- Layer 2: A fragments from LDS scratch ----------------
    v16h a2[4];
    #pragma unroll
    for (int kk = 0; kk < 4; ++kk) {
        const _Float16* ar = hs + lr * 128 + kk * 32 + half * 16;
        v8h lo = *(const v8h*)(ar);
        v8h hi = *(const v8h*)(ar + 8);
        a2[kk] = __builtin_shufflevector(lo, hi,
                 0, 1, 2, 3, 4, 5, 6, 7, 8, 9, 10, 11, 12, 13, 14, 15);
    }

    // Layer 2 + fused layer-3 head: p[r] += silu(h2[row][col]) * w3[col],
    // accumulated directly in the C/D register layout (no LDS round-trip).
    v8f p;
    #pragma unroll
    for (int r = 0; r < 8; ++r) p[r] = 0.0f;

    #pragma unroll
    for (int ct = 0; ct < 8; ++ct) {
        const float bias = b2s[ct * 16 + lr];
        const float w3c  = w3s[ct * 16 + lr];
        v8f acc;
        #pragma unroll
        for (int i = 0; i < 8; ++i) acc[i] = bias;
        const _Float16* wb = &W2h[(ct * 16 + lr) * 128 + half * 16];
        #pragma unroll
        for (int kk = 0; kk < 4; ++kk) {
            v16h b = *(const v16h*)(wb + kk * 32);
            acc = __builtin_amdgcn_wmma_f32_16x16x32_f16(
                false, a2[kk], false, b, (short)0, acc, false, false);
        }
        #pragma unroll
        for (int r = 0; r < 8; ++r)
            p[r] += silu_f(acc[r]) * w3c;
    }

    // Butterfly-reduce across the 16 columns held by each lane half.
    #pragma unroll
    for (int d = 1; d < 16; d <<= 1) {
        #pragma unroll
        for (int r = 0; r < 8; ++r)
            p[r] += __shfl_xor(p[r], d, 32);
    }
    // Lane lr (0..7) of each half owns row half*8 + lr.
    if (lr < 8) {
        float myv = p[0];
        #pragma unroll
        for (int r = 1; r < 8; ++r) myv = (lr == r) ? p[r] : myv;
        const int myrow = nodeBase + half * 8 + lr;
        if (myrow < N) {
            atomicAdd(&iso[batch[myrow]], myv + b3v);
        }
    }

    // ---------------- l2 path: 5 rows per node dotted with w_l2 ----------------
    // Dominant HBM traffic (512 MB): lane-striped float4 loads, 512B per wave-load.
    v4f wl2 = *(const v4f*)(wl2g + lane * 4);
    for (int i = 0; i < 16; ++i) {
        const int n = nodeBase + i;
        if (n >= N) break;
        const int g = batch[n];
        const float* base = ne + ((size_t)n * 9 + 4) * 128 + lane * 4;
        // Prefetch next node's l2 rows (global_prefetch_b8).
        if (n + 1 < N)
            __builtin_prefetch(ne + ((size_t)(n + 1) * 9 + 4) * 128 + lane * 4, 0, 0);
        #pragma unroll
        for (int m = 0; m < 5; ++m) {
            v4f x = *(const v4f*)(base + m * 128);
            float pd = x[0] * wl2[0] + x[1] * wl2[1] + x[2] * wl2[2] + x[3] * wl2[3];
            pd += __shfl_xor(pd, 1, 32);
            pd += __shfl_xor(pd, 2, 32);
            pd += __shfl_xor(pd, 4, 32);
            pd += __shfl_xor(pd, 8, 32);
            pd += __shfl_xor(pd, 16, 32);
            if (lane == 0) atomicAdd(&aniso[g * 5 + m], pd);
        }
    }
}

// Per-graph: scale by 1/natoms, apply fixed dec @ CG, emit (3,3).
__global__ void finalize_kernel(const float* __restrict__ iso,
                                const float* __restrict__ aniso,
                                const int*   __restrict__ natoms,
                                float* __restrict__ out, int G)
{
    int g = blockIdx.x * blockDim.x + threadIdx.x;
    if (g >= G) return;
    const float inv = 1.0f / (float)natoms[g];
    const float s  = iso[g] * inv;
    const float a0 = aniso[g * 5 + 0] * inv;
    const float a1 = aniso[g * 5 + 1] * inv;
    const float a2 = aniso[g * 5 + 2] * inv;
    const float a3 = aniso[g * 5 + 3] * inv;
    const float a4 = aniso[g * 5 + 4] * inv;
    const float S2 = 0.70710678118654752f;   // 2^-0.5
    const float S3 = 0.57735026918962576f;   // 3^-0.5
    const float S6 = 0.40824829046386302f;   // 6^-0.5
    float* o = out + (size_t)g * 9;
    o[0] = s * S3 - a2 * S6 - a4 * S2;
    o[1] = a3 * S2;
    o[2] = a0 * S2;
    o[3] = a3 * S2;
    o[4] = s * S3 + 2.0f * a2 * S6;
    o[5] = a1 * S2;
    o[6] = a0 * S2;
    o[7] = a1 * S2;
    o[8] = s * S3 - a2 * S6 + a4 * S2;
}

extern "C" void kernel_launch(void* const* d_in, const int* in_sizes, int n_in,
                              void* d_out, int out_size, void* d_ws, size_t ws_size,
                              hipStream_t stream)
{
    const float* ne     = (const float*)d_in[0];
    const float* W1     = (const float*)d_in[1];
    const float* b1     = (const float*)d_in[2];
    const float* W2     = (const float*)d_in[3];
    const float* b2     = (const float*)d_in[4];
    const float* W3     = (const float*)d_in[5];
    const float* b3     = (const float*)d_in[6];
    const float* wl2    = (const float*)d_in[7];
    const int*   batch  = (const int*)d_in[8];
    const int*   natoms = (const int*)d_in[9];

    const int N = in_sizes[0] / (9 * 128);
    const int G = in_sizes[9];

    float* iso   = (float*)d_ws;       // G floats
    float* aniso = iso + G;            // G*5 floats

    const int zn = 6 * G;
    zero_ws_kernel<<<(zn + 255) / 256, 256, 0, stream>>>(iso, zn);

    const int nblocks = (N + 127) / 128;
    mlp_reduce_kernel<<<nblocks, 256, 0, stream>>>(
        ne, W1, b1, W2, b2, W3, b3, wl2, batch, iso, aniso, N);

    finalize_kernel<<<(G + 255) / 256, 256, 0, stream>>>(
        iso, aniso, natoms, (float*)d_out, G);
}
